// MoELayer_56676388438162
// MI455X (gfx1250) — compile-verified, matrix operand order checked by reading, not run on hardware
//
#include <hip/hip_runtime.h>
#include <hip/hip_bf16.h>

// ---------------------------------------------------------------------------
// MoE (Switch top-1) for MI455X / gfx1250, wave32 + v_wmma_f32_16x16x32_bf16.
//   K1: zero per-expert counters.
//   K2: router (logits, argmax, top-1 prob, per-expert token lists).
//   K3: weight repack f32 -> bf16 in WMMA-fragment order (one HBM pass,
//       amortizes conversion that round-1 paid ~64x per weight element).
//   K4: grouped fused FFN; hot loop = ds_load_b128 x2 + global_load_b128 x2
//       + v_wmma per 16K FLOPs. Weights (75MB bf16) are L2-resident (192MB L2).
//   K4b: fallback gather/convert FFN if ws_size too small for repack.
// ---------------------------------------------------------------------------

#define D_DIM 768
#define F_DIM 3072
#define N_EXP 8
#define TILE_M 16
#define F_CHUNK 128
#define KB1 (D_DIM / 32)   // 24 k-blocks for GEMM1
#define NT1 (F_DIM / 16)   // 192 n-tiles for GEMM1
#define KB2 (F_DIM / 32)   // 96 k-blocks for GEMM2
#define NT2 (D_DIM / 16)   // 48 n-tiles for GEMM2

typedef __attribute__((ext_vector_type(16))) __bf16 v16bf;
typedef __attribute__((ext_vector_type(8)))  float  v8f;

__device__ __forceinline__ unsigned short bf16bits(float f) {
    unsigned u = __float_as_uint(f);
    u += 0x7FFFu + ((u >> 16) & 1u);          // round-to-nearest-even
    return (unsigned short)(u >> 16);
}
__device__ __forceinline__ __bf16 f2bf(float f) {
    return __builtin_bit_cast(__bf16, bf16bits(f));
}

// ISA 7.12.2: 16-bit operand fragment, lane half 'hi', element j -> K index.
__device__ __forceinline__ int kmap(int j, int hi) {
    return (j & 7) + ((j & 8) << 1) + (hi << 3);
}

// ------------------------------- init --------------------------------------
__global__ void moe_init_kernel(int* __restrict__ counts) {
    if (threadIdx.x < N_EXP) counts[threadIdx.x] = 0;
}

// ------------------------------ router -------------------------------------
__global__ __launch_bounds__(256) void moe_router_kernel(
    const float* __restrict__ X, const float* __restrict__ Wr,
    float* __restrict__ logits_out, int* __restrict__ idx_out,
    float* __restrict__ probs_ws, int* __restrict__ counts_ws,
    int* __restrict__ toks_ws, int NT)
{
    const int wv   = threadIdx.x >> 5;
    const int lane = threadIdx.x & 31;
    const int token = blockIdx.x * 8 + wv;
    if (token >= NT) return;

    float acc[N_EXP];
#pragma unroll
    for (int e = 0; e < N_EXP; ++e) acc[e] = 0.f;

    const float* xp = X + (size_t)token * D_DIM;
    for (int d = lane; d < D_DIM; d += 32) {
        float x = xp[d];
#pragma unroll
        for (int e = 0; e < N_EXP; ++e) acc[e] += x * Wr[d * N_EXP + e];
    }
#pragma unroll
    for (int off = 16; off > 0; off >>= 1) {
#pragma unroll
        for (int e = 0; e < N_EXP; ++e) acc[e] += __shfl_xor(acc[e], off, 32);
    }
    if (lane == 0) {
        float m = acc[0]; int ei = 0;
#pragma unroll
        for (int e = 1; e < N_EXP; ++e)
            if (acc[e] > m) { m = acc[e]; ei = e; }   // first-max on ties
        float s = 0.f;
#pragma unroll
        for (int e = 0; e < N_EXP; ++e) s += __expf(acc[e] - m);
        float p = 1.0f / s;
#pragma unroll
        for (int e = 0; e < N_EXP; ++e)
            logits_out[(size_t)token * N_EXP + e] = acc[e];
        idx_out[token]  = ei;
        probs_ws[token] = p;
        int slot = atomicAdd(&counts_ws[ei], 1);
        toks_ws[(size_t)ei * NT + slot] = token;
    }
}

// --------------------------- weight repack ---------------------------------
// One thread per (fragment, lane): reads a 16-deep K column (coalesced across
// the 16 lanes of a fragment half), converts to bf16, writes one contiguous
// 32B lane-slice. Fragment = 32x16 (KxN) operand tile for v_wmma bf16.
// blockIdx.y: 0 -> W1 [E,D,F], 1 -> W2 [E,F,D].
__global__ __launch_bounds__(256) void moe_repack_kernel(
    const float* __restrict__ W1, const float* __restrict__ W2,
    unsigned short* __restrict__ w1bf, unsigned short* __restrict__ w2bf)
{
    const int gid  = blockIdx.x * 256 + threadIdx.x;
    const int lane = gid & 31;
    const int hi   = lane >> 4;
    const int n    = lane & 15;
    const int frag = gid >> 5;

    const float* src;
    unsigned short* dst;
    size_t stride;
    if (blockIdx.y == 0) {                 // W1: frag = (e*KB1 + kb)*NT1 + nt
        const int nt = frag % NT1; int r2 = frag / NT1;
        const int kb = r2 % KB1;  const int e = r2 / KB1;
        src = W1 + (size_t)e * D_DIM * F_DIM + (size_t)(kb * 32) * F_DIM
                 + nt * 16 + n;
        stride = F_DIM;
        dst = w1bf + ((size_t)frag * 32 + lane) * 16;
    } else {                               // W2: frag = (e*KB2 + kb)*NT2 + nt
        const int nt = frag % NT2; int r2 = frag / NT2;
        const int kb = r2 % KB2;  const int e = r2 / KB2;
        src = W2 + (size_t)e * F_DIM * D_DIM + (size_t)(kb * 32) * D_DIM
                 + nt * 16 + n;
        stride = D_DIM;
        dst = w2bf + ((size_t)frag * 32 + lane) * 16;
    }

    unsigned short pk[16];
#pragma unroll
    for (int j = 0; j < 16; ++j)
        pk[j] = bf16bits(src[(size_t)kmap(j, hi) * stride]);

    unsigned u[8];
#pragma unroll
    for (int i = 0; i < 8; ++i)
        u[i] = (unsigned)pk[2 * i] | ((unsigned)pk[2 * i + 1] << 16);
    uint4* d4 = (uint4*)dst;
    d4[0] = make_uint4(u[0], u[1], u[2], u[3]);
    d4[1] = make_uint4(u[4], u[5], u[6], u[7]);
}

// ----------------------- fused FFN (repacked weights) ----------------------
// Block = 256 threads (8 waves) per (expert, 16-token tile).
// Per F-chunk of 128: GEMM1 (wave -> one 16x16 F-subtile, K over D), relu ->
// fragment-order LDS, GEMM2 (wave -> 6 of 48 D-subtiles, K over chunk).
__global__ __launch_bounds__(256) void moe_ffn_wmma_kernel(
    const float* __restrict__ X,
    const unsigned short* __restrict__ w1bf, const float* __restrict__ b1,
    const unsigned short* __restrict__ w2bf, const float* __restrict__ b2,
    const int* __restrict__ counts, const int* __restrict__ toks,
    const float* __restrict__ probs, float* __restrict__ out, int NT)
{
    const int e    = blockIdx.y;
    const int base = blockIdx.x * TILE_M;
    const int cnt  = counts[e];
    if (base >= cnt) return;
    const int cntTile = (cnt - base < TILE_M) ? (cnt - base) : TILE_M;

    // Fragment-order staging: [k-block][lane][16 elems] -> 32B/lane loads.
    __shared__ __align__(32) unsigned short XsF[KB1][32][16];   // 24.0 KB
    __shared__ __align__(32) unsigned short HsF[4][32][16];     //  4.0 KB
    __shared__ int    s_tok[TILE_M];
    __shared__ float  s_prob[TILE_M];

    const int tid  = threadIdx.x;
    const int wv   = tid >> 5;
    const int lane = tid & 31;
    const int nl   = lane & 15;
    const int hi   = lane >> 4;

    if (tid < TILE_M) {
        int t = (tid < cntTile) ? toks[(size_t)e * NT + base + tid] : -1;
        s_tok[tid]  = t;
        s_prob[tid] = (t >= 0) ? probs[t] : 0.f;
    }
    __syncthreads();

    // Stage token tile as bf16 A-fragments (vectorized float4 reads).
    for (int pos = tid; pos < KB1 * 32; pos += 256) {
        const int kb = pos >> 5, ln = pos & 31;
        const int tok = s_tok[ln & 15];
        const int khi = (ln >> 4) << 3;               // hi*8
        unsigned u[8];
        if (tok >= 0) {
            const float* sp = X + (size_t)tok * D_DIM + kb * 32 + khi;
            float4 f0 = *(const float4*)(sp);
            float4 f1 = *(const float4*)(sp + 4);
            float4 f2 = *(const float4*)(sp + 16);
            float4 f3 = *(const float4*)(sp + 20);
            float v[16] = { f0.x, f0.y, f0.z, f0.w, f1.x, f1.y, f1.z, f1.w,
                            f2.x, f2.y, f2.z, f2.w, f3.x, f3.y, f3.z, f3.w };
#pragma unroll
            for (int i = 0; i < 8; ++i)
                u[i] = (unsigned)bf16bits(v[2 * i]) |
                       ((unsigned)bf16bits(v[2 * i + 1]) << 16);
        } else {
#pragma unroll
            for (int i = 0; i < 8; ++i) u[i] = 0u;
        }
        unsigned* dst = (unsigned*)&XsF[kb][ln][0];
#pragma unroll
        for (int i = 0; i < 8; ++i) dst[i] = u[i];
    }
    __syncthreads();

    v8f acc2[6];
#pragma unroll
    for (int t = 0; t < 6; ++t) acc2[t] = v8f{};

    const unsigned short* w1e = w1bf + (size_t)e * D_DIM * F_DIM;
    const unsigned short* w2e = w2bf + (size_t)e * F_DIM * D_DIM;

    // GEMM1 output (C layout: N=nl, M=r+8*hi) -> GEMM2 A-fragment remap:
    const int kb2w = wv >> 1;                         // dest k-block in chunk
    const int hi2  = (nl >> 3) & 1;
    const int jj   = (nl & 7) | ((wv & 1) << 3);

    for (int fc = 0; fc < F_DIM; fc += F_CHUNK) {
        const int ntbase = fc >> 4;                   // first F n-tile of chunk
        const int kbbase = fc >> 5;                   // first F k-block (GEMM2)

        // Prefetch next chunk's fragments into L2 (global_prefetch_b8).
        if (fc + F_CHUNK < F_DIM) {
            int f1i = tid % (KB1 * 8);                // 192 fragments (GEMM1)
            __builtin_prefetch(w1e + (((size_t)(f1i / 8) * NT1
                               + (ntbase + 8 + (f1i & 7))) * 32) * 16, 0, 1);
            int f2i = tid % (4 * NT2);                // 192 fragments (GEMM2)
            __builtin_prefetch(w2e + (((size_t)(kbbase + 4 + (f2i / NT2)) * NT2
                               + (f2i % NT2)) * 32) * 16, 0, 1);
        }

        // ---- GEMM1: H[16x128] = X[16x768] * W1[:, chunk] ------------------
        const int nt1 = ntbase + wv;                  // this wave's F n-tile
        const unsigned short* bp1 = w1e + ((size_t)nt1 * 32 + lane) * 16;
        v8f acc1 = v8f{};
#pragma unroll 4
        for (int kb = 0; kb < KB1; ++kb) {
            v16bf a = *(const v16bf*)&XsF[kb][lane][0];
            v16bf b = *(const v16bf*)bp1;
            bp1 += (size_t)NT1 * 32 * 16;
            acc1 = __builtin_amdgcn_wmma_f32_16x16x32_bf16(
                       false, a, false, b, (short)0, acc1, false, false);
        }
        const float bias1 = b1[e * F_DIM + nt1 * 16 + nl];
#pragma unroll
        for (int r = 0; r < 8; ++r) {
            const int m = r + (hi << 3);
            float h = acc1[r] + bias1;
            HsF[kb2w][(hi2 << 4) | m][jj] = bf16bits(h > 0.f ? h : 0.f);
        }
        __syncthreads();

        // ---- GEMM2: out[16x768] += H[16x128] * W2[chunk, :] ---------------
#pragma unroll
        for (int t = 0; t < 6; ++t) {
            const int nt2 = wv * 6 + t;               // D n-tile 0..47
#pragma unroll
            for (int kb = 0; kb < 4; ++kb) {
                v16bf a = *(const v16bf*)&HsF[kb][lane][0];
                v16bf b = *(const v16bf*)(w2e +
                          (((size_t)(kbbase + kb) * NT2 + nt2) * 32 + lane) * 16);
                acc2[t] = __builtin_amdgcn_wmma_f32_16x16x32_bf16(
                              false, a, false, b, (short)0, acc2[t], false, false);
            }
        }
        __syncthreads();   // HsF reusable next chunk
    }

    // ---- epilogue: +b2, * router prob, scatter token rows -----------------
#pragma unroll
    for (int t = 0; t < 6; ++t) {
        const int col = ((wv * 6 + t) << 4);
        const float bias2 = b2[e * D_DIM + col + nl];
#pragma unroll
        for (int r = 0; r < 8; ++r) {
            const int m = r + (hi << 3);
            if (m < cntTile) {
                float v = (acc2[t][r] + bias2) * s_prob[m];
                out[(size_t)s_tok[m] * D_DIM + col + nl] = v;
            }
        }
    }
}

// ------------- fallback FFN (direct gather/convert, small ws) --------------
__global__ __launch_bounds__(256) void moe_ffn_fallback_kernel(
    const float* __restrict__ X,  const float* __restrict__ W1,
    const float* __restrict__ b1, const float* __restrict__ W2,
    const float* __restrict__ b2, const int* __restrict__ counts,
    const int* __restrict__ toks, const float* __restrict__ probs,
    float* __restrict__ out, int NT)
{
    const int e    = blockIdx.y;
    const int base = blockIdx.x * TILE_M;
    const int cnt  = counts[e];
    if (base >= cnt) return;
    const int cntTile = (cnt - base < TILE_M) ? (cnt - base) : TILE_M;

    __shared__ __bf16 Xs[TILE_M][D_DIM + 4];
    __shared__ __bf16 Hs[TILE_M][F_CHUNK + 4];
    __shared__ int    s_tok[TILE_M];
    __shared__ float  s_prob[TILE_M];

    const int tid  = threadIdx.x;
    const int wv   = tid >> 5;
    const int lane = tid & 31;
    const int nl   = lane & 15;
    const int hi   = lane >> 4;

    if (tid < TILE_M) {
        int t = (tid < cntTile) ? toks[(size_t)e * NT + base + tid] : -1;
        s_tok[tid]  = t;
        s_prob[tid] = (t >= 0) ? probs[t] : 0.f;
    }
    __syncthreads();
    for (int idx = tid; idx < TILE_M * D_DIM; idx += 256) {
        int row = idx / D_DIM, col = idx - row * D_DIM;
        int t = s_tok[row];
        Xs[row][col] = f2bf((t >= 0) ? X[(size_t)t * D_DIM + col] : 0.f);
    }
    __syncthreads();

    v8f acc2[6];
#pragma unroll
    for (int t = 0; t < 6; ++t) acc2[t] = v8f{};

    const float* W1e = W1 + (size_t)e * D_DIM * F_DIM;
    const float* W2e = W2 + (size_t)e * F_DIM * D_DIM;

    for (int fc = 0; fc < F_DIM; fc += F_CHUNK) {
        const int fcol = fc + (wv << 4);
        const float* B1p = W1e + fcol + nl;
        v8f acc1 = v8f{};
        for (int ks = 0; ks < D_DIM; ks += 32) {
            v16bf a, b;
#pragma unroll
            for (int j = 0; j < 16; ++j) {
                int k = ks + kmap(j, hi);
                a[j] = __builtin_bit_cast(__bf16, ((unsigned short(*)[D_DIM + 4])Xs)[nl][k]);
                b[j] = f2bf(B1p[(size_t)k * F_DIM]);
            }
            acc1 = __builtin_amdgcn_wmma_f32_16x16x32_bf16(
                       false, a, false, b, (short)0, acc1, false, false);
        }
        const float bias1 = b1[e * F_DIM + fcol + nl];
#pragma unroll
        for (int r = 0; r < 8; ++r) {
            float h = acc1[r] + bias1;
            Hs[r + (hi << 3)][(wv << 4) + nl] = f2bf(h > 0.f ? h : 0.f);
        }
        __syncthreads();
#pragma unroll
        for (int t = 0; t < 6; ++t) {
            const int col = ((wv * 6 + t) << 4);
            const float* B2p = W2e + (size_t)fc * D_DIM + col + nl;
#pragma unroll
            for (int ks = 0; ks < F_CHUNK; ks += 32) {
                v16bf a, b;
#pragma unroll
                for (int j = 0; j < 16; ++j) {
                    int k = ks + kmap(j, hi);
                    a[j] = __builtin_bit_cast(__bf16, ((unsigned short(*)[F_CHUNK + 4])Hs)[nl][k]);
                    b[j] = f2bf(B2p[(size_t)k * D_DIM]);
                }
                acc2[t] = __builtin_amdgcn_wmma_f32_16x16x32_bf16(
                              false, a, false, b, (short)0, acc2[t], false, false);
            }
        }
        __syncthreads();
    }
#pragma unroll
    for (int t = 0; t < 6; ++t) {
        const int col = ((wv * 6 + t) << 4);
        const float bias2 = b2[e * D_DIM + col + nl];
#pragma unroll
        for (int r = 0; r < 8; ++r) {
            const int m = r + (hi << 3);
            if (m < cntTile)
                out[(size_t)s_tok[m] * D_DIM + col + nl] =
                    (acc2[t][r] + bias2) * s_prob[m];
        }
    }
}

// ------------------------------- launch ------------------------------------
extern "C" void kernel_launch(void* const* d_in, const int* in_sizes, int n_in,
                              void* d_out, int out_size, void* d_ws, size_t ws_size,
                              hipStream_t stream) {
    const float* X  = (const float*)d_in[0];
    const float* Wr = (const float*)d_in[1];
    const float* W1 = (const float*)d_in[2];
    const float* b1 = (const float*)d_in[3];
    const float* W2 = (const float*)d_in[4];
    const float* b2 = (const float*)d_in[5];

    const int NT = in_sizes[0] / D_DIM;        // 8192 tokens

    float* out    = (float*)d_out;
    float* logits = out + (size_t)NT * D_DIM;
    int*   eidx   = (int*)(logits + (size_t)NT * N_EXP);

    char*  ws     = (char*)d_ws;
    int*   counts = (int*)ws;
    int*   toks   = (int*)(ws + 32);
    float* probs  = (float*)(ws + 32 + (size_t)N_EXP * NT * 4);
    size_t off    = 32 + (size_t)N_EXP * NT * 4 + (size_t)NT * 4;
    off = (off + 255) & ~(size_t)255;
    unsigned short* w1bf = (unsigned short*)(ws + off);
    unsigned short* w2bf = w1bf + (size_t)N_EXP * D_DIM * F_DIM;
    const size_t need = off + 2ull * (size_t)N_EXP * D_DIM * F_DIM * 2ull; // ~76 MB

    moe_init_kernel<<<1, 32, 0, stream>>>(counts);
    moe_router_kernel<<<NT / 8, 256, 0, stream>>>(X, Wr, logits, eidx,
                                                  probs, counts, toks, NT);
    dim3 grid((NT + TILE_M - 1) / TILE_M, N_EXP);
    if (ws_size >= need) {
        const int nfrag_threads = N_EXP * KB1 * NT1 * 32;  // == N_EXP*KB2*NT2*32
        dim3 rg(nfrag_threads / 256, 2);
        moe_repack_kernel<<<rg, 256, 0, stream>>>(W1, W2, w1bf, w2bf);
        moe_ffn_wmma_kernel<<<grid, 256, 0, stream>>>(X, w1bf, b1, w2bf, b2,
                                                      counts, toks, probs, out, NT);
    } else {
        moe_ffn_fallback_kernel<<<grid, 256, 0, stream>>>(X, W1, b1, W2, b2,
                                                          counts, toks, probs, out, NT);
    }
}
